// MSDeformAttnTransformerEncoder_27144193311280
// MI455X (gfx1250) — compile-verified, matrix operand order checked by reading, not run on hardware
//
#include <hip/hip_runtime.h>

// ---------------------------------------------------------------------------
// MSDeformAttn encoder for MI455X (gfx1250, wave32, WMMA).
// All dense projections run through a WMMA f16 (f32-accumulate) tiled GEMM;
// sampling / softmax / layernorm are VALU kernels.
// ---------------------------------------------------------------------------

#define LQ    18720
#define BSZ   2
#define CCH   256
#define FFD   1024
#define NLAY  6
#define NLEVS 4
#define NHEAD 8
#define NPT   4
#define HDIM  32
#define MTOT  (BSZ * LQ)      // 37440 rows

typedef __attribute__((ext_vector_type(16))) _Float16 v16h;
typedef __attribute__((ext_vector_type(2)))  __fp16   fp16x2;  // cvt_pkrtz result type
typedef __attribute__((ext_vector_type(8)))  float    v8f;
typedef __attribute__((ext_vector_type(4)))  int      i4;

union FragH { v16h h; i4 q[2]; };

// ---------------------------------------------------------------------------
// WMMA GEMM:  Out[M,N] = act(A_f16[M,K] * W_f32[K,N] + bias[N])
// Block tile 128x128, K-step 32. 256 threads = 8 waves in a 2(M) x 4(N) grid;
// each wave computes a 64x32 patch = 4x2 v_wmma_f32_16x16x32_f16 tiles
// (8 WMMAs per 12 ds_load_b128 per K-step).
// Requires: N % 128 == 0, K % 32 == 0 (all shapes here satisfy this).
// ---------------------------------------------------------------------------
#define BM 128
#define BN 128
#define BK 32

__global__ __launch_bounds__(256) void gemm_bias_act(
    const _Float16* __restrict__ A, const float* __restrict__ W,
    const float* __restrict__ bias, float* __restrict__ Out,
    int Mrows, int N, int K, int do_relu)
{
    __shared__ _Float16 As[BM * BK];   // [m][k], 8 KB
    __shared__ _Float16 Bs[BN * BK];   // [n][k] (transposed on store), 8 KB

    const int tid   = threadIdx.x;
    const int lane  = tid & 31;
    const int lhalf = lane >> 4;       // 0: lanes 0-15, 1: lanes 16-31
    const int r16   = lane & 15;
    const int wid   = tid >> 5;
    const int wm    = wid & 1;         // wave M position (0..1), 64 rows each
    const int wn    = wid >> 1;        // wave N position (0..3), 32 cols each
    const int row0  = blockIdx.x * BM;
    const int col0  = blockIdx.y * BN;

    v8f zero = {};
    v8f acc[4][2];
    #pragma unroll
    for (int mt = 0; mt < 4; ++mt) { acc[mt][0] = zero; acc[mt][1] = zero; }

    for (int k0 = 0; k0 < K; k0 += BK) {
        // ---- stage A tile (128x32 f16): 2 threads per row, 16 f16 each ----
        {
            const int r  = tid >> 1;
            const int cb = (tid & 1) * 16;
            const int grow = row0 + r;
            i4 d0 = {}, d1 = {};
            if (grow < Mrows) {
                const i4* src = (const i4*)(A + (size_t)grow * K + k0 + cb);
                d0 = src[0];
                d1 = src[1];
                if (k0 + BK < K)   // speculative prefetch of next K tile
                    __builtin_prefetch(A + (size_t)grow * K + k0 + BK + cb, 0, 0);
            }
            *(i4*)&As[r * BK + cb]     = d0;
            *(i4*)&As[r * BK + cb + 8] = d1;
        }
        // ---- stage W tile (32x128 f32 -> f16, stored [n][k]) ------------
        // Each thread owns one column n and 16 consecutive k's; global dword
        // loads are coalesced across adjacent lanes (consecutive n), pairs
        // packed with v_cvt_pk_rtz_f16_f32, two ds_store_b128 per thread.
        {
            const int nl = tid & 127;          // 0..127
            const int kb = (tid >> 7) * 16;    // 0 or 16
            const float* src = W + (size_t)(k0 + kb) * N + col0 + nl;
            union { fp16x2 h[8]; i4 q[2]; } pk;
            #pragma unroll
            for (int j = 0; j < 8; ++j) {
                const float a0 = src[(size_t)(2 * j)     * N];
                const float a1 = src[(size_t)(2 * j + 1) * N];
                pk.h[j] = __builtin_amdgcn_cvt_pkrtz(a0, a1);
            }
            *(i4*)&Bs[nl * BK + kb]     = pk.q[0];
            *(i4*)&Bs[nl * BK + kb + 8] = pk.q[1];
        }
        __syncthreads();

        // ---- fragment loads: contiguous 16B per half-lane, matching the
        //      16-bit A/B layout (lane<16: K 0..7 & 16..23; lane>=16: 8..15 & 24..31)
        FragH af[4], bf[2];
        #pragma unroll
        for (int mt = 0; mt < 4; ++mt) {
            const int rl = wm * 64 + mt * 16 + r16;
            af[mt].q[0] = *(const i4*)&As[rl * BK + lhalf * 8];
            af[mt].q[1] = *(const i4*)&As[rl * BK + 16 + lhalf * 8];
        }
        #pragma unroll
        for (int nt = 0; nt < 2; ++nt) {
            const int nc = wn * 32 + nt * 16 + r16;
            bf[nt].q[0] = *(const i4*)&Bs[nc * BK + lhalf * 8];
            bf[nt].q[1] = *(const i4*)&Bs[nc * BK + 16 + lhalf * 8];
        }
        #pragma unroll
        for (int mt = 0; mt < 4; ++mt)
            #pragma unroll
            for (int nt = 0; nt < 2; ++nt)
                acc[mt][nt] = __builtin_amdgcn_wmma_f32_16x16x32_f16(
                    false, af[mt].h, false, bf[nt].h,
                    (short)0, acc[mt][nt], false, false);
        __syncthreads();
    }

    // ---- epilogue: C/D layout => VGPR j: row = j + 8*lhalf, col = r16 ----
    #pragma unroll
    for (int mt = 0; mt < 4; ++mt) {
        #pragma unroll
        for (int nt = 0; nt < 2; ++nt) {
            const int rowb = row0 + wm * 64 + mt * 16 + lhalf * 8;
            const int col  = col0 + wn * 32 + nt * 16 + r16;
            const float bv = bias[col];
            #pragma unroll
            for (int j = 0; j < 8; ++j) {
                const int row = rowb + j;
                if (row < Mrows) {
                    float v = acc[mt][nt][j] + bv;
                    if (do_relu) v = fmaxf(v, 0.0f);
                    Out[(size_t)row * N + col] = v;
                }
            }
        }
    }
}

// ---------------------------------------------------------------------------
// Build x[b,q,c] (into d_out) and pos_embed[b,q,c] (+ level embedding)
// ---------------------------------------------------------------------------
__global__ __launch_bounds__(256) void init_x_pos(
    const float* __restrict__ f0, const float* __restrict__ f1,
    const float* __restrict__ f2, const float* __restrict__ f3,
    const float* __restrict__ p0, const float* __restrict__ p1,
    const float* __restrict__ p2, const float* __restrict__ p3,
    const float* __restrict__ le, float* __restrict__ x,
    float* __restrict__ pose)
{
    long i = (long)blockIdx.x * blockDim.x + threadIdx.x;
    if (i >= (long)MTOT * CCH) return;
    const int  c = (int)(i % CCH);
    const long m = i / CCH;
    const int  b = (int)(m / LQ);
    const int  q = (int)(m % LQ);
    const float *f, *p; int lev, s;
    if      (q < 16384) { f = f0; p = p0; lev = 0; s = 0;     }
    else if (q < 18432) { f = f1; p = p1; lev = 1; s = 16384; }
    else if (q < 18688) { f = f2; p = p2; lev = 2; s = 18432; }
    else                { f = f3; p = p3; lev = 3; s = 18688; }
    const int nsz = (16 >> lev) * (32 >> lev) * (32 >> lev);
    const long src = ((long)b * CCH + c) * nsz + (q - s);
    x[i]    = f[src];
    pose[i] = p[src] + le[lev * CCH + c];
}

// normalized voxel-center reference points, z-major flatten
__global__ void build_ref(float* __restrict__ ref)
{
    int q = blockIdx.x * blockDim.x + threadIdx.x;
    if (q >= LQ) return;
    int lev, s;
    if      (q < 16384) { lev = 0; s = 0;     }
    else if (q < 18432) { lev = 1; s = 16384; }
    else if (q < 18688) { lev = 2; s = 18432; }
    else                { lev = 3; s = 18688; }
    const int D = 16 >> lev, H = 32 >> lev, W = 32 >> lev;
    const int n = q - s;
    const int z = n / (H * W), r = n % (H * W), y = r / W, xx = r % W;
    ref[q * 3 + 0] = (xx + 0.5f) / W;
    ref[q * 3 + 1] = (y  + 0.5f) / H;
    ref[q * 3 + 2] = (z  + 0.5f) / D;
}

// q16 = f16(x + pos), x16 = f16(x); two elements per thread, packed stores
__global__ __launch_bounds__(256) void prep_q16(
    const float* __restrict__ x, const float* __restrict__ pose,
    _Float16* __restrict__ q16, _Float16* __restrict__ x16)
{
    long i = (long)blockIdx.x * blockDim.x + threadIdx.x;   // pair index
    if (i >= (long)MTOT * CCH / 2) return;
    const float2 xv = ((const float2*)x)[i];
    const float2 pv = ((const float2*)pose)[i];
    union { fp16x2 h; unsigned u; } a, b;
    a.h = __builtin_amdgcn_cvt_pkrtz(xv.x + pv.x, xv.y + pv.y);
    b.h = __builtin_amdgcn_cvt_pkrtz(xv.x, xv.y);
    ((unsigned*)q16)[i] = a.u;
    ((unsigned*)x16)[i] = b.u;
}

// packed f32 -> f16 conversion, two elements per thread
__global__ __launch_bounds__(256) void cvt16(
    const float* __restrict__ in, _Float16* __restrict__ out, long n2)
{
    long i = (long)blockIdx.x * blockDim.x + threadIdx.x;   // pair index
    if (i >= n2) return;
    const float2 f = ((const float2*)in)[i];
    union { fp16x2 h; unsigned u; } pk;
    pk.h = __builtin_amdgcn_cvt_pkrtz(f.x, f.y);
    ((unsigned*)out)[i] = pk.u;
}

// softmax over groups of 16 (NLEV*NP), one thread per (m,h); in place
__global__ __launch_bounds__(256) void softmax16(float* __restrict__ a, int rows)
{
    int i = blockIdx.x * blockDim.x + threadIdx.x;
    if (i >= rows) return;
    float* p = a + (long)i * 16;
    float mx = p[0];
    #pragma unroll
    for (int j = 1; j < 16; ++j) mx = fmaxf(mx, p[j]);
    float sum = 0.0f, e[16];
    #pragma unroll
    for (int j = 0; j < 16; ++j) { e[j] = __expf(p[j] - mx); sum += e[j]; }
    const float inv = 1.0f / sum;
    #pragma unroll
    for (int j = 0; j < 16; ++j) p[j] = e[j] * inv;
}

// ---------------------------------------------------------------------------
// Deformable trilinear sampling: one wave per (b,q,h), lane = head channel.
// grid_sample(bilinear, zeros, align_corners=False) semantics.
// ---------------------------------------------------------------------------
__global__ __launch_bounds__(256) void deform_sample(
    const float* __restrict__ val, const float* __restrict__ off,
    const float* __restrict__ attn, const float* __restrict__ ref,
    float* __restrict__ out)
{
    const int w    = blockIdx.x * 8 + (threadIdx.x >> 5);
    const int lane = threadIdx.x & 31;
    if (w >= MTOT * NHEAD) return;
    const int  h = w % NHEAD;
    const long m = w / NHEAD;
    const int  b = (int)(m / LQ);
    const int  q = (int)(m % LQ);

    const float rx = ref[q * 3 + 0], ry = ref[q * 3 + 1], rz = ref[q * 3 + 2];
    const float* offp = off  + m * (long)(NHEAD * NLEVS * NPT * 3) + h * (NLEVS * NPT * 3);
    const float* ap   = attn + m * (long)(NHEAD * NLEVS * NPT)     + h * (NLEVS * NPT);

    const int starts[4] = {0, 16384, 18432, 18688};
    float acc = 0.0f;
    #pragma unroll
    for (int lev = 0; lev < NLEVS; ++lev) {
        const int D = 16 >> lev, H = 32 >> lev, W = 32 >> lev;
        const float* vbase = val + ((long)b * LQ + starts[lev]) * CCH + h * HDIM + lane;
        #pragma unroll
        for (int p = 0; p < NPT; ++p) {
            const int   sp = lev * NPT + p;
            const float aw = ap[sp];
            const float gx = (rx + offp[sp * 3 + 0] / W) * W - 0.5f;
            const float gy = (ry + offp[sp * 3 + 1] / H) * H - 0.5f;
            const float gz = (rz + offp[sp * 3 + 2] / D) * D - 0.5f;
            const float fx0 = floorf(gx), fy0 = floorf(gy), fz0 = floorf(gz);
            const int x0 = (int)fx0, y0 = (int)fy0, z0 = (int)fz0;
            const float fx = gx - fx0, fy = gy - fy0, fz = gz - fz0;
            float samp = 0.0f;
            #pragma unroll
            for (int dz = 0; dz < 2; ++dz) {
                const int zi = z0 + dz;
                const float wz = dz ? fz : 1.0f - fz;
                const bool okz = (zi >= 0) && (zi < D);
                const int zc = min(max(zi, 0), D - 1);
                #pragma unroll
                for (int dy = 0; dy < 2; ++dy) {
                    const int yi = y0 + dy;
                    const float wy = dy ? fy : 1.0f - fy;
                    const bool oky = okz && (yi >= 0) && (yi < H);
                    const int yc = min(max(yi, 0), H - 1);
                    #pragma unroll
                    for (int dx = 0; dx < 2; ++dx) {
                        const int xi = x0 + dx;
                        const float wx = dx ? fx : 1.0f - fx;
                        const bool ok = oky && (xi >= 0) && (xi < W);
                        const int xc = min(max(xi, 0), W - 1);
                        const float wgt = wz * wy * wx * (ok ? 1.0f : 0.0f);
                        const float v = vbase[(long)((zc * H + yc) * W + xc) * CCH];
                        samp += wgt * v;
                    }
                }
            }
            acc += aw * samp;
        }
    }
    out[m * CCH + h * HDIM + lane] = acc;
}

// ---------------------------------------------------------------------------
// x = LayerNorm(x + t) * g + b   — one wave per row, wave32 shfl reduction
// ---------------------------------------------------------------------------
__global__ __launch_bounds__(256) void ln_residual(
    float* __restrict__ x, const float* __restrict__ tb,
    const float* __restrict__ g, const float* __restrict__ bb)
{
    const int row  = blockIdx.x * 8 + (threadIdx.x >> 5);
    const int lane = threadIdx.x & 31;
    if (row >= MTOT) return;
    float v[8], s = 0.0f, s2 = 0.0f;
    #pragma unroll
    for (int k = 0; k < 8; ++k) {
        const int c = k * 32 + lane;
        const float u = x[(long)row * CCH + c] + tb[(long)row * CCH + c];
        v[k] = u; s += u; s2 += u * u;
    }
    #pragma unroll
    for (int o = 16; o > 0; o >>= 1) {
        s  += __shfl_xor(s,  o, 32);
        s2 += __shfl_xor(s2, o, 32);
    }
    const float mean = s  * (1.0f / CCH);
    const float var  = s2 * (1.0f / CCH) - mean * mean;
    const float inv  = rsqrtf(var + 1e-5f);
    #pragma unroll
    for (int k = 0; k < 8; ++k) {
        const int c = k * 32 + lane;
        x[(long)row * CCH + c] = (v[k] - mean) * inv * g[c] + bb[c];
    }
}

// ---------------------------------------------------------------------------
extern "C" void kernel_launch(void* const* d_in, const int* in_sizes, int n_in,
                              void* d_out, int out_size, void* d_ws, size_t ws_size,
                              hipStream_t stream)
{
    (void)in_sizes; (void)n_in; (void)out_size; (void)ws_size;

    const float* f0 = (const float*)d_in[0];
    const float* p0 = (const float*)d_in[1];
    const float* f1 = (const float*)d_in[2];
    const float* p1 = (const float*)d_in[3];
    const float* f2 = (const float*)d_in[4];
    const float* p2 = (const float*)d_in[5];
    const float* f3 = (const float*)d_in[6];
    const float* p3 = (const float*)d_in[7];
    const float* le  = (const float*)d_in[8];
    const float* Wso = (const float*)d_in[9];
    const float* bso = (const float*)d_in[10];
    const float* Waw = (const float*)d_in[11];
    const float* baw = (const float*)d_in[12];
    const float* Wv  = (const float*)d_in[13];
    const float* bv  = (const float*)d_in[14];
    const float* Wo  = (const float*)d_in[15];
    const float* bo  = (const float*)d_in[16];
    const float* g1  = (const float*)d_in[17];
    const float* b1  = (const float*)d_in[18];
    const float* Wl1 = (const float*)d_in[19];
    const float* bl1 = (const float*)d_in[20];
    const float* Wl2 = (const float*)d_in[21];
    const float* bl2 = (const float*)d_in[22];
    const float* g2  = (const float*)d_in[23];
    const float* b2  = (const float*)d_in[24];

    float* x = (float*)d_out;                 // x lives in d_out throughout

    char* ws = (char*)d_ws;
    size_t o = 0;
    auto take = [&](size_t bytes) -> char* {
        char* p = ws + o;
        o += (bytes + 255) & ~(size_t)255;
        return p;
    };
    float*    pose   = (float*)   take((size_t)MTOT * CCH * 4);
    float*    ref    = (float*)   take((size_t)LQ * 3 * 4);
    _Float16* q16    = (_Float16*)take((size_t)MTOT * CCH * 2);
    _Float16* x16    = (_Float16*)take((size_t)MTOT * CCH * 2);
    float*    offb   = (float*)   take((size_t)MTOT * 384 * 4);
    float*    attn   = (float*)   take((size_t)MTOT * 128 * 4);
    float*    valb   = (float*)   take((size_t)MTOT * CCH * 4);
    float*    samp   = (float*)   take((size_t)MTOT * CCH * 4);
    _Float16* samp16 = (_Float16*)take((size_t)MTOT * CCH * 2);
    float*    tbuf   = (float*)   take((size_t)MTOT * CCH * 4);
    float*    hbuf   = (float*)   take((size_t)MTOT * FFD * 4);
    _Float16* h16    = (_Float16*)take((size_t)MTOT * FFD * 2);

    const int  nMC  = MTOT * CCH;                       // 9,584,640
    const long nMF  = (long)MTOT * FFD;                 // 38,338,560
    const int  gMC  = (nMC + 255) / 256;
    const int  gMC2 = (nMC / 2 + 255) / 256;            // pair-granular grids
    const int  gMF2 = (int)((nMF / 2 + 255) / 256);
    const int  gM128 = (MTOT + 127) / 128;

    init_x_pos<<<gMC, 256, 0, stream>>>(f0, f1, f2, f3, p0, p1, p2, p3, le, x, pose);
    build_ref<<<(LQ + 255) / 256, 256, 0, stream>>>(ref);

    for (int l = 0; l < NLAY; ++l) {
        prep_q16<<<gMC2, 256, 0, stream>>>(x, pose, q16, x16);

        gemm_bias_act<<<dim3(gM128, 384 / 128), 256, 0, stream>>>(
            q16, Wso + (size_t)l * CCH * 384, bso + (size_t)l * 384,
            offb, MTOT, 384, CCH, 0);
        gemm_bias_act<<<dim3(gM128, 128 / 128), 256, 0, stream>>>(
            q16, Waw + (size_t)l * CCH * 128, baw + (size_t)l * 128,
            attn, MTOT, 128, CCH, 0);
        softmax16<<<(MTOT * NHEAD + 255) / 256, 256, 0, stream>>>(attn, MTOT * NHEAD);

        gemm_bias_act<<<dim3(gM128, CCH / 128), 256, 0, stream>>>(
            x16, Wv + (size_t)l * CCH * CCH, bv + (size_t)l * CCH,
            valb, MTOT, CCH, CCH, 0);

        deform_sample<<<(MTOT * NHEAD + 7) / 8, 256, 0, stream>>>(valb, offb, attn, ref, samp);

        cvt16<<<gMC2, 256, 0, stream>>>(samp, samp16, (long)nMC / 2);
        gemm_bias_act<<<dim3(gM128, CCH / 128), 256, 0, stream>>>(
            samp16, Wo + (size_t)l * CCH * CCH, bo + (size_t)l * CCH,
            tbuf, MTOT, CCH, CCH, 0);
        ln_residual<<<(MTOT + 7) / 8, 256, 0, stream>>>(x, tbuf, g1 + (size_t)l * CCH, b1 + (size_t)l * CCH);

        cvt16<<<gMC2, 256, 0, stream>>>(x, x16, (long)nMC / 2);
        gemm_bias_act<<<dim3(gM128, FFD / 128), 256, 0, stream>>>(
            x16, Wl1 + (size_t)l * CCH * FFD, bl1 + (size_t)l * FFD,
            hbuf, MTOT, FFD, CCH, 1);
        cvt16<<<gMF2, 256, 0, stream>>>(hbuf, h16, nMF / 2);
        gemm_bias_act<<<dim3(gM128, CCH / 128), 256, 0, stream>>>(
            h16, Wl2 + (size_t)l * FFD * CCH, bl2 + (size_t)l * CCH,
            tbuf, MTOT, CCH, FFD, 0);
        ln_residual<<<(MTOT + 7) / 8, 256, 0, stream>>>(x, tbuf, g2 + (size_t)l * CCH, b2 + (size_t)l * CCH);
    }
}
